// SlayerMM_88364657148602
// MI455X (gfx1250) — matrix operand order repaired and checked by reading, not compiled, not run on hardware
//
#include <hip/hip_runtime.h>
#include <hip/hip_bf16.h>

// SLAYER VT_SNN forward on gfx1250 (MI455X).
//
// Roofline: dense/psp commute (both linear ops on different axes), so dense
// runs FIRST on the 6300-channel vis path (5.0 GMAC psp -> 8 MMAC). The
// remaining dominant cost is one streaming read of spike_vis
// (32*6300*325*4B = 262 MB f32 -> ~11 us at 23.3 TB/s). Everything else is
// noise, so the GEMMs use v_wmma_f32_16x16x32_f16 (0/1 spikes are exact in
// f16) with a branch-free inner loop so the 262 MB stream is pipelined.

typedef __attribute__((ext_vector_type(16))) _Float16 v16h;
typedef __attribute__((ext_vector_type(8)))  float    v8f;

#define SL_THETA 10.0f
constexpr int SL_T = 325;                 // T_SAMPLE (compile-time: enables
                                          // immediate-offset B loads)

// ---------------------------------------------------------------------------
// Pack W [O,I] f32 into WMMA A-operand layout, f16, zero-padded:
//   Apack[((mt*Kc + c)*32 + lane)*16 + j] = W[mt*16 + (lane&15),
//                                            c*32 + 16*(lane>>4) + j]
// One thread per packed half element. (contiguous k order 16*(lane>>4)+j, not the ISA table order; B loads match)
// ---------------------------------------------------------------------------
__global__ void slayer_pack_wA(const float* __restrict__ W,
                               _Float16* __restrict__ A,
                               int O, int I, int Kc, int total)
{
    const int idx = blockIdx.x * blockDim.x + threadIdx.x;
    if (idx >= total) return;
    const int j  = idx & 15;
    const int l  = (idx >> 4) & 31;
    const int tc = idx >> 9;              // mt*Kc + c
    const int c  = tc % Kc;
    const int mt = tc / Kc;
    const int o  = mt * 16 + (l & 15);
    const int g  = l >> 4;
    const int k  = c * 32 + 16 * g + j;
    const float v = (o < O && k < I) ? W[o * I + k] : 0.0f;
    A[idx] = (_Float16)v;
}

// ---------------------------------------------------------------------------
// WMMA GEMM: U[b,o,t] = sum_i W[o,i] * X[b,i,t]
// One wave per 16x16 (o,t) tile per batch element. A from packed f16 buffer
// (contiguous 32B/lane/chunk). B loaded from f32 X with per-lane column t
// (wave-coalesced across lanes) and 16 immediate-offset dword loads per chunk.
// Branch-free inner loop -> EXEC all-ones at the WMMA, loads pipelined.
// ---------------------------------------------------------------------------
__global__ void slayer_wmma_dense(const _Float16* __restrict__ Apack,
                                  const float* __restrict__ X, // [B,I,T]
                                  float* __restrict__ U,       // [B,O,T]
                                  int O, int I, int Kc)
{
    const int lane = threadIdx.x & 31;
    const int g    = lane >> 4;
    const int mn   = lane & 15;
    constexpr int nTileT = (SL_T + 15) / 16;   // 21
    const int mtile = blockIdx.x / nTileT;
    const int ntile = blockIdx.x % nTileT;
    const int b     = blockIdx.y;
    const int t0    = ntile * 16;
    // Clamp t: out-of-range columns compute garbage that is never stored.
    const int t     = (t0 + mn < SL_T) ? (t0 + mn) : (SL_T - 1);

    const _Float16* __restrict__ Ap =
        Apack + ((size_t)mtile * Kc * 32 + lane) * 16;
    const float* __restrict__ Xb = X + (size_t)b * I * SL_T + t;

    v8f acc = {};
    const int Kfull = I >> 5;                  // chunks with no k masking
    int c = 0;
    for (; c < Kfull; ++c) {
        const v16h av = *(const v16h*)(Ap + (size_t)c * 512);
        const float* __restrict__ xp = Xb + (size_t)(c * 32 + 16 * g) * SL_T;
        v16h bv;
        #pragma unroll
        for (int j = 0; j < 16; ++j)           // 16 loads, immediate offsets
            bv[j] = (_Float16)xp[j * SL_T];
        acc = __builtin_amdgcn_wmma_f32_16x16x32_f16(
                false, av, false, bv, (short)0, acc, false, false);
    }
    if (c < Kc) {                              // masked epilogue chunk
        const v16h av = *(const v16h*)(Ap + (size_t)c * 512);
        v16h bv;
        #pragma unroll
        for (int j = 0; j < 16; ++j) {
            const int kb = c * 32 + 16 * g + j;
            const int kc = (kb < I) ? kb : (I - 1);        // clamped address
            const float xv = Xb[(size_t)kc * SL_T];        // unconditional load
            bv[j] = (_Float16)((kb < I) ? xv : 0.0f);      // v_cndmask, no branch
        }
        acc = __builtin_amdgcn_wmma_f32_16x16x32_f16(
                false, av, false, bv, (short)0, acc, false, false);
    }

    #pragma unroll
    for (int r = 0; r < 8; ++r) {              // C/D: VGPR r, group g -> row r+8g
        const int oo = mtile * 16 + r + 8 * g;
        const int tt = t0 + mn;
        if (oo < O && tt < SL_T)
            U[((size_t)b * O + oo) * SL_T + tt] = acc[r];
    }
}

// ---------------------------------------------------------------------------
// psp: causal temporal conv per (b,n) row; y[t] = sum_{k<=t} ker[k]*u[t-k]
// One block per row; row + kernel staged in LDS. TS == 1.0.
// ---------------------------------------------------------------------------
__global__ void slayer_psp(const float* __restrict__ U,
                           const float* __restrict__ kern,
                           float* __restrict__ Y, int K)
{
    __shared__ float srow[352];
    __shared__ float sker[352];
    const int row = blockIdx.x;
    const float* __restrict__ u = U + (size_t)row * SL_T;
    for (int i = threadIdx.x; i < SL_T; i += blockDim.x) srow[i] = u[i];
    for (int i = threadIdx.x; i < K;    i += blockDim.x) sker[i] = kern[i];
    __syncthreads();
    for (int t = threadIdx.x; t < SL_T; t += blockDim.x) {
        float acc = 0.0f;
        const int kmax = (K - 1 < t) ? (K - 1) : t;
        for (int k = 0; k <= kmax; ++k)
            acc = fmaf(sker[k], srow[t - k], acc);
        Y[(size_t)row * SL_T + t] = acc;
    }
}

// ---------------------------------------------------------------------------
// spike: SLAYER threshold scan with additive refractory response.
// One thread per (b,n) sequence; circular refractory buffer in LDS
// (64 threads x 33 floats, conflict-free). Writes into concat channel layout.
// ---------------------------------------------------------------------------
__global__ void slayer_spike(const float* __restrict__ P,   // [rows, T]
                             const float* __restrict__ refk, int Kref,
                             float* __restrict__ S,
                             int N, int chanOff, int chanTot, int rows)
{
    __shared__ float sref[33];
    __shared__ float buf[64 * 33];
    const int L = Kref - 1;                   // refractory buffer length (<=32)
    for (int i = threadIdx.x; i < L; i += blockDim.x) sref[i] = refk[1 + i];
    __syncthreads();

    const int idx = blockIdx.x * blockDim.x + threadIdx.x;
    if (idx >= rows) return;
    const int b = idx / N;
    const int n = idx - b * N;
    const float* __restrict__ p = P + (size_t)idx * SL_T;
    float* __restrict__ out = S + ((size_t)b * chanTot + chanOff + n) * SL_T;

    float* myb = &buf[threadIdx.x * 33];
    for (int j = 0; j < L; ++j) myb[j] = 0.0f;

    int head = 0;
    for (int t = 0; t < SL_T; ++t) {
        const float v = p[t] + ((L > 0) ? myb[head] : 0.0f);
        const float s = (v >= SL_THETA) ? 1.0f : 0.0f;     // 1/TS, TS==1
        if (L > 0) {
            myb[head] = 0.0f;                 // consumed slot -> new tail
            head = (head + 1 == L) ? 0 : head + 1;
            if (s != 0.0f) {
                for (int j = 0; j < L; ++j) {
                    int q = head + j; if (q >= L) q -= L;
                    myb[q] += sref[j];
                }
            }
        }
        out[t] = s;
    }
}

// ---------------------------------------------------------------------------
extern "C" void kernel_launch(void* const* d_in, const int* in_sizes, int n_in,
                              void* d_out, int out_size, void* d_ws, size_t ws_size,
                              hipStream_t stream) {
    (void)n_in; (void)out_size; (void)ws_size;
    const float* spike_tact = (const float*)d_in[0]; // [32, 156, 325]
    const float* spike_vis  = (const float*)d_in[1]; // [32, 6300, 325]
    const float* w_tact     = (const float*)d_in[2]; // [50, 156]
    const float* w_vis      = (const float*)d_in[3]; // [10, 6300]
    const float* w_out      = (const float*)d_in[4]; // [20, 60]
    const float* srm        = (const float*)d_in[5]; // alpha kernel, len ~77
    const float* refk       = (const float*)d_in[6]; // refractory kernel, ~17

    int Ksrm = in_sizes[5]; if (Ksrm > 352) Ksrm = 352;
    int Kref = in_sizes[6]; if (Kref > 33)  Kref = 33;

    const int B = 32, T = SL_T;
    const int nTileT = (T + 15) / 16;                 // 21

    // GEMM geometry: {O, I} -> Mtiles, Kchunks
    const int MT_tact = 4, KC_tact = (156 + 31) / 32;   // 4, 5
    const int MT_vis  = 1, KC_vis  = (6300 + 31) / 32;  // 1, 197
    const int MT_out  = 2, KC_out  = (60 + 31) / 32;    // 2, 2

    // workspace (floats): s_all[B*60*T] | ubuf[B*50*T] | pbuf[B*50*T] | Apacks
    float* s_all = (float*)d_ws;
    float* ubuf  = s_all + (size_t)B * 60 * T;
    float* pbuf  = ubuf  + (size_t)B * 50 * T;
    _Float16* apT = (_Float16*)(pbuf + (size_t)B * 50 * T);
    const int nT = MT_tact * KC_tact * 512;           // packed half counts
    const int nV = MT_vis  * KC_vis  * 512;
    const int nO = MT_out  * KC_out  * 512;
    _Float16* apV = apT + nT;
    _Float16* apO = apV + nV;

    // --- pack all weight matrices into WMMA A layout (tiny)
    slayer_pack_wA<<<(nT + 255) / 256, 256, 0, stream>>>(w_tact, apT, 50, 156,  KC_tact, nT);
    slayer_pack_wA<<<(nV + 255) / 256, 256, 0, stream>>>(w_vis,  apV, 10, 6300, KC_vis,  nV);
    slayer_pack_wA<<<(nO + 255) / 256, 256, 0, stream>>>(w_out,  apO, 20, 60,   KC_out,  nO);

    // --- tact encoder: dense(156->50) -> psp -> spike -> s_all[:, 0:50, :]
    slayer_wmma_dense<<<dim3(MT_tact * nTileT, B), 32, 0, stream>>>(
        apT, spike_tact, ubuf, 50, 156, KC_tact);
    slayer_psp<<<B * 50, 128, 0, stream>>>(ubuf, srm, pbuf, Ksrm);
    slayer_spike<<<(B * 50 + 63) / 64, 64, 0, stream>>>(
        pbuf, refk, Kref, s_all, 50, /*chanOff=*/0, /*chanTot=*/60, B * 50);

    // --- vis encoder: dense(6300->10) -> psp -> spike -> s_all[:, 50:60, :]
    slayer_wmma_dense<<<dim3(MT_vis * nTileT, B), 32, 0, stream>>>(
        apV, spike_vis, ubuf, 10, 6300, KC_vis);
    slayer_psp<<<B * 10, 128, 0, stream>>>(ubuf, srm, pbuf, Ksrm);
    slayer_spike<<<(B * 10 + 63) / 64, 64, 0, stream>>>(
        pbuf, refk, Kref, s_all, 10, /*chanOff=*/50, /*chanTot=*/60, B * 10);

    // --- head: dense(60->20) -> psp -> spike -> d_out
    slayer_wmma_dense<<<dim3(MT_out * nTileT, B), 32, 0, stream>>>(
        apO, s_all, ubuf, 20, 60, KC_out);
    slayer_psp<<<B * 20, 128, 0, stream>>>(ubuf, srm, pbuf, Ksrm);
    slayer_spike<<<(B * 20 + 63) / 64, 64, 0, stream>>>(
        pbuf, refk, Kref, (float*)d_out, 20, /*chanOff=*/0, /*chanTot=*/20, B * 20);
}
